// SS2D_50345606644066
// MI455X (gfx1250) — matrix operand order repaired and checked by reading, not compiled, not run on hardware
//
#include <hip/hip_runtime.h>

// ---------------- problem constants ----------------
#define TOK     32768      // B * L = 8 * 4096 tokens
#define DMODEL  384
#define DINNER  768
#define CONVDIM 800
#define DPROJ   1580
#define DPROJP  1600       // padded so N tiles of 64 divide evenly
#define NHEADS  12
#define HEADDIM 64
#define DSTATE  16
#define NCHUNK  64
#define BATCH   8
#define HW      64

typedef __attribute__((ext_vector_type(16))) __bf16 bf16x16;
typedef __attribute__((ext_vector_type(8)))  __bf16 bf16x8;
typedef __attribute__((ext_vector_type(8)))  float  f32x8;

__device__ __forceinline__ __bf16 f2bf(float f) {
  unsigned u = __builtin_bit_cast(unsigned, f);
  u += 0x7FFFu + ((u >> 16) & 1u);           // round-to-nearest-even
  unsigned short s = (unsigned short)(u >> 16);
  return __builtin_bit_cast(__bf16, s);
}

// ---- WMMA bf16 fragment loads from LDS, per ISA 7.12.2 layouts (wave32) ----
// A (16x32): lane holds row M=lane%16; needs K runs {0..7,16..23} (lanes<16)
// or {8..15,24..31} (lanes>=16)  -> two 16B contiguous LDS loads.
__device__ __forceinline__ bf16x16 ldsFragA(const __bf16* row, int lane) {
  const int o = (lane >= 16) ? 8 : 0;
  bf16x8 lo = *(const bf16x8*)(row + o);
  bf16x8 hi = *(const bf16x8*)(row + o + 16);
  return __builtin_shufflevector(lo, hi, 0,1,2,3,4,5,6,7,8,9,10,11,12,13,14,15);
}
// B (32x16): lane holds col N=lane%16; needs K runs {0..7,16..23} (lanes<16)
// or {8..15,24..31} (lanes>=16) -> two 16B loads when tile stored [N][K].
__device__ __forceinline__ bf16x16 ldsFragB(const __bf16* row, int lane) {
  const int o = (lane >= 16) ? 8 : 0;
  bf16x8 lo = *(const bf16x8*)(row + o);
  bf16x8 hi = *(const bf16x8*)(row + o + 16);
  return __builtin_shufflevector(lo, hi, 0,1,2,3,4,5,6,7,8,9,10,11,12,13,14,15);
}

// ---------------- weight prep ----------------
__global__ void k_prep_win(const float* __restrict__ Win, __bf16* __restrict__ WinBf) {
  int idx = blockIdx.x * 256 + threadIdx.x;
  if (idx >= DMODEL * DPROJP) return;
  int r = idx / DPROJP, c = idx % DPROJP;
  float v = (c < DPROJ) ? Win[(size_t)r * DPROJ + c] : 0.f;
  WinBf[idx] = f2bf(v);
}

__global__ void k_prep_wout(const float* __restrict__ Wout, __bf16* __restrict__ WoutBf) {
  int idx = blockIdx.x * 256 + threadIdx.x;
  if (idx >= DINNER * DMODEL) return;
  WoutBf[idx] = f2bf(Wout[idx]);
}

// ---------------- GEMM1: zx[32768x1600] = x[32768x384] @ WinBf[384x1600] ----------------
__global__ __launch_bounds__(256) void k_gemm_inproj(const float* __restrict__ X,
                                                     const __bf16* __restrict__ Wb,
                                                     float* __restrict__ Y) {
  __shared__ __align__(16) __bf16 sA[128][40];   // [M][K], 80B row stride (16B mult.)
  __shared__ __align__(16) __bf16 sBt[64][40];   // [N][K] transposed weight tile
  const int tid = threadIdx.x, lane = tid & 31, wave = tid >> 5;
  const int row0 = blockIdx.y * 128, col0 = blockIdx.x * 64;
  const int m0 = wave * 16, mr = lane & 15;
  f32x8 acc[4];
  for (int nt = 0; nt < 4; ++nt)
    for (int j = 0; j < 8; ++j) acc[nt][j] = 0.f;

  for (int kt = 0; kt < DMODEL / 32; ++kt) {
    const int k0 = kt * 32;
    // stage A: float4 loads, fp32 -> bf16
    for (int i = 0; i < 4; ++i) {
      int idx = tid + i * 256;                 // 1024 float4 chunks
      int r = idx >> 3, c4 = idx & 7;
      const float4 v = *(const float4*)&X[(size_t)(row0 + r) * DMODEL + k0 + c4 * 4];
      sA[r][c4 * 4 + 0] = f2bf(v.x);
      sA[r][c4 * 4 + 1] = f2bf(v.y);
      sA[r][c4 * 4 + 2] = f2bf(v.z);
      sA[r][c4 * 4 + 3] = f2bf(v.w);
    }
    // stage B: b128 loads, transposed store [N][K]
    {
      int r = tid >> 3, c8 = tid & 7;          // 256 chunks of 8 bf16
      bf16x8 wv = *(const bf16x8*)&Wb[(size_t)(k0 + r) * DPROJP + col0 + c8 * 8];
      for (int j = 0; j < 8; ++j) sBt[c8 * 8 + j][r] = wv[j];
    }
    if (kt + 1 < DMODEL / 32)                  // -> global_prefetch_b8
      __builtin_prefetch(&X[(size_t)(row0 + (tid >> 1)) * DMODEL + k0 + 32], 0, 1);
    __syncthreads();

    bf16x16 af = ldsFragA(&sA[m0 + mr][0], lane);
    for (int nt = 0; nt < 4; ++nt) {
      bf16x16 bfr = ldsFragB(&sBt[nt * 16 + mr][0], lane);
      acc[nt] = __builtin_amdgcn_wmma_f32_16x16x32_bf16(false, af, false, bfr,
                                                        (short)0, acc[nt], false, false);
    }
    __syncthreads();
  }
  const int mrow = m0 + 8 * (lane >> 4);
  for (int nt = 0; nt < 4; ++nt)
    for (int j = 0; j < 8; ++j)
      Y[(size_t)(row0 + mrow + j) * DPROJP + col0 + nt * 16 + mr] = acc[nt][j];
}

// ---------------- dt = softplus(raw + dt_bias) ----------------
__global__ void k_dt(const float* __restrict__ zx, const float* __restrict__ dt_bias,
                     float* __restrict__ dtBuf) {
  int idx = blockIdx.x * 256 + threadIdx.x;
  if (idx >= TOK * NHEADS) return;
  int t = idx / NHEADS, h = idx % NHEADS;
  float xv = zx[(size_t)t * DPROJP + DINNER + CONVDIM + h] + dt_bias[h];
  dtBuf[idx] = (xv > 20.f) ? xv : log1pf(expf(xv));
}

// ---------------- depthwise 3x3 conv (SAME) + bias + SiLU ----------------
__global__ void k_conv(const float* __restrict__ zx, const float* __restrict__ Wc,
                       const float* __restrict__ bc, float* __restrict__ xconv) {
  int idx = blockIdx.x * 256 + threadIdx.x;
  if (idx >= BATCH * HW * HW * CONVDIM) return;
  int ch = idx % CONVDIM;
  int r = idx / CONVDIM;
  int xx = r % HW; r /= HW;
  int yy = r % HW; int b = r / HW;
  float acc = bc[ch];
  for (int ky = 0; ky < 3; ++ky) {
    int iy = yy + ky - 1;
    if (iy < 0 || iy >= HW) continue;
    for (int kx = 0; kx < 3; ++kx) {
      int ix = xx + kx - 1;
      if (ix < 0 || ix >= HW) continue;
      acc += zx[(size_t)(b * 4096 + iy * HW + ix) * DPROJP + DINNER + ch] *
             Wc[(ky * 3 + kx) * CONVDIM + ch];
    }
  }
  xconv[idx] = acc / (1.f + __expf(-acc));    // SiLU
}

// ---------------- SSD intra-chunk: Y_diag, chunk states, cumsum(A) ----------------
__global__ __launch_bounds__(128) void k_ssd_intra(
    const float* __restrict__ xconv, const float* __restrict__ dtBuf,
    const float* __restrict__ A_log, const float* __restrict__ D_param,
    float* __restrict__ Yssd, float* __restrict__ chunkStates,
    float* __restrict__ acumBuf) {
  // bf16 staging tiles laid out so every WMMA operand row is K-contiguous:
  __shared__ __align__(16) __bf16 sCb[64][40];   // C rows, K=n (16..31 zero pad)
  __shared__ __align__(16) __bf16 sBb[64][40];   // B rows, K=n (16..31 zero pad)
  __shared__ __align__(16) __bf16 sXsT[64][72];  // Xs^T: [p][l]
  __shared__ __align__(16) __bf16 sGb[64][72];   // masked decay G: [l][s]
  __shared__ __align__(16) __bf16 sBdT[16][72];  // (decay*B)^T: [n][l]
  __shared__ float sB[64][17];
  __shared__ float sDt[64];
  __shared__ float sAcum[64];
  const int tid = threadIdx.x, lane = tid & 31, wave = tid >> 5;
  const int c = blockIdx.x, h = blockIdx.y, b = blockIdx.z;
  const size_t tb = (size_t)b * 4096 + (size_t)c * 64;
  const float Ah = -__expf(A_log[h]);

  if (tid < 64) sDt[tid] = dtBuf[(tb + tid) * NHEADS + h];
  __syncthreads();
  if (tid == 0) {                 // serial cumsum of a[l] = Ah*dt[l]
    float s = 0.f;
    for (int l = 0; l < 64; ++l) { s += Ah * sDt[l]; sAcum[l] = s; }
  }
  for (int i = 0; i < 32; ++i) {  // dt-scaled X, transposed bf16
    int idx = tid + i * 128;
    int l = idx >> 6, p = idx & 63;
    sXsT[p][l] = f2bf(xconv[(tb + l) * CONVDIM + h * HEADDIM + p] * sDt[l]);
  }
  for (int i = 0; i < 8; ++i) {
    int idx = tid + i * 128;
    int l = idx >> 4, n = idx & 15;
    float bv = xconv[(tb + l) * CONVDIM + DINNER + n];
    float cv = xconv[(tb + l) * CONVDIM + DINNER + DSTATE + n];
    sB[l][n] = bv;
    sBb[l][n] = f2bf(bv);  sBb[l][16 + n] = f2bf(0.f);
    sCb[l][n] = f2bf(cv);  sCb[l][16 + n] = f2bf(0.f);
  }
  __syncthreads();

  // decayed B^T (needs cumsum): sBdT[n][l] = B[l][n]*exp(acum[63]-acum[l])
  const float T = sAcum[63];
  for (int i = 0; i < 8; ++i) {
    int idx = tid + i * 128;       // 16*64 = 1024
    int n = idx >> 6, l = idx & 63;
    sBdT[n][l] = f2bf(sB[l][n] * __expf(T - sAcum[l]));
  }

  const int m0 = wave * 16, mr = lane & 15;
  // G = C * B^T  (64x64, K=16 zero-padded to WMMA K=32), masked segsum decay
  {
    bf16x16 af = ldsFragA(&sCb[m0 + mr][0], lane);
    for (int nt = 0; nt < 4; ++nt) {
      bf16x16 bfr = ldsFragB(&sBb[nt * 16 + mr][0], lane);
      f32x8 acc;
      for (int j = 0; j < 8; ++j) acc[j] = 0.f;
      acc = __builtin_amdgcn_wmma_f32_16x16x32_bf16(false, af, false, bfr,
                                                    (short)0, acc, false, false);
      for (int j = 0; j < 8; ++j) {
        int l = m0 + j + 8 * (lane >> 4);
        int s = nt * 16 + mr;
        sGb[l][s] = (s <= l) ? f2bf(acc[j] * __expf(sAcum[l] - sAcum[s])) : f2bf(0.f);
      }
    }
  }
  __syncthreads();   // sGb + sBdT visible to all waves

  // Y_diag = G @ Xs  (64x64, K=64) + D * x_inner
  {
    f32x8 acc[4];
    for (int nt = 0; nt < 4; ++nt)
      for (int j = 0; j < 8; ++j) acc[nt][j] = 0.f;
    for (int kt = 0; kt < 2; ++kt) {
      bf16x16 af = ldsFragA(&sGb[m0 + mr][kt * 32], lane);
      for (int nt = 0; nt < 4; ++nt) {
        bf16x16 bfr = ldsFragB(&sXsT[nt * 16 + mr][kt * 32], lane);
        acc[nt] = __builtin_amdgcn_wmma_f32_16x16x32_bf16(false, af, false, bfr,
                                                          (short)0, acc[nt], false, false);
      }
    }
    const float Dh = D_param[h];
    for (int nt = 0; nt < 4; ++nt)
      for (int j = 0; j < 8; ++j) {
        int l = m0 + j + 8 * (lane >> 4);
        int p = nt * 16 + mr;
        float xin = xconv[(tb + l) * CONVDIM + h * HEADDIM + p];
        Yssd[(tb + l) * DINNER + h * HEADDIM + p] = acc[nt][j] + Dh * xin;
      }
  }

  // chunk states: S[p][n] = sum_l Xs[l][p] * exp(acum[63]-acum[l]) * B[l][n]
  {
    f32x8 acc;
    for (int j = 0; j < 8; ++j) acc[j] = 0.f;
    for (int kt = 0; kt < 2; ++kt) {
      bf16x16 af  = ldsFragA(&sXsT[m0 + mr][kt * 32], lane);   // A = Xs^T (p x l)
      bf16x16 bfr = ldsFragB(&sBdT[mr][kt * 32], lane);        // B = decay*B (l x n)
      acc = __builtin_amdgcn_wmma_f32_16x16x32_bf16(false, af, false, bfr,
                                                    (short)0, acc, false, false);
    }
    const size_t bch = (size_t)(b * NCHUNK + c) * NHEADS + h;
    for (int j = 0; j < 8; ++j) {
      int p = m0 + j + 8 * (lane >> 4);
      chunkStates[(bch * 64 + p) * 16 + mr] = acc[j];
    }
    if (tid < 64) acumBuf[bch * 64 + tid] = sAcum[tid];
  }
}

// ---------------- SSD inter-chunk scan (fp32, precision-critical) ----------------
__global__ __launch_bounds__(256) void k_ssd_inter(
    const float* __restrict__ xconv, const float* __restrict__ acumBuf,
    const float* __restrict__ chunkStates, float* __restrict__ Yssd) {
  __shared__ float sS[64][17];    // running state S[p][n]
  __shared__ float sC[64][17];
  __shared__ float sAc[64];
  const int tid = threadIdx.x;
  const int h = blockIdx.x, b = blockIdx.y;
  for (int i = 0; i < 4; ++i) {
    int idx = tid + i * 256;
    sS[idx >> 4][idx & 15] = 0.f;
  }
  __syncthreads();
  for (int c = 0; c < NCHUNK; ++c) {
    const size_t tb = (size_t)b * 4096 + (size_t)c * 64;
    const size_t bch = (size_t)(b * NCHUNK + c) * NHEADS + h;
    for (int i = 0; i < 4; ++i) {
      int idx = tid + i * 256;
      int l = idx >> 4, n = idx & 15;
      sC[l][n] = xconv[(tb + l) * CONVDIM + DINNER + DSTATE + n];
    }
    if (tid < 64) sAc[tid] = acumBuf[bch * 64 + tid];
    __syncthreads();
    // Y_off[l][p] = exp(acum[l]) * sum_n C[l][n] * S[p][n]   (S BEFORE update)
    for (int i = 0; i < 16; ++i) {
      int idx = tid + i * 256;
      int l = idx >> 6, p = idx & 63;
      float acc = 0.f;
      for (int n = 0; n < DSTATE; ++n) acc += sC[l][n] * sS[p][n];
      Yssd[(tb + l) * DINNER + h * HEADDIM + p] += acc * __expf(sAc[l]);
    }
    __syncthreads();
    const float eT = __expf(sAc[63]);
    for (int i = 0; i < 4; ++i) {
      int idx = tid + i * 256;
      int p = idx >> 4, n = idx & 15;
      sS[p][n] = sS[p][n] * eT + chunkStates[(bch * 64 + p) * 16 + n];
    }
    __syncthreads();
  }
}

// ---------------- LayerNorm over D_INNER + SiLU(z) gate, emit bf16 ----------------
__global__ __launch_bounds__(256) void k_norm_gate(
    const float* __restrict__ Yssd, const float* __restrict__ zx,
    const float* __restrict__ gamma, const float* __restrict__ beta,
    __bf16* __restrict__ yg) {
  __shared__ float r1[256], r2[256];
  const int t = blockIdx.x, tid = threadIdx.x;
  float v[3];
  float s1 = 0.f, s2 = 0.f;
  for (int i = 0; i < 3; ++i) {
    float y = Yssd[(size_t)t * DINNER + tid + i * 256];
    v[i] = y; s1 += y; s2 += y * y;
  }
  r1[tid] = s1; r2[tid] = s2;
  __syncthreads();
  for (int off = 128; off > 0; off >>= 1) {
    if (tid < off) { r1[tid] += r1[tid + off]; r2[tid] += r2[tid + off]; }
    __syncthreads();
  }
  const float mu = r1[0] * (1.f / 768.f);
  const float var = r2[0] * (1.f / 768.f) - mu * mu;
  const float rstd = rsqrtf(var + 1e-5f);
  for (int i = 0; i < 3; ++i) {
    int cch = tid + i * 256;
    float z = zx[(size_t)t * DPROJP + cch];
    float g = z / (1.f + __expf(-z));
    float o = (v[i] - mu) * rstd * gamma[cch] + beta[cch];
    yg[(size_t)t * DINNER + cch] = f2bf(o * g);
  }
}

// ---------------- GEMM2: out[32768x384] = yg[32768x768] @ WoutBf[768x384] ----------------
__global__ __launch_bounds__(256) void k_gemm_outproj(const __bf16* __restrict__ Ag,
                                                      const __bf16* __restrict__ Wb,
                                                      float* __restrict__ Y) {
  __shared__ __align__(16) __bf16 sA[128][40];
  __shared__ __align__(16) __bf16 sBt[64][40];
  const int tid = threadIdx.x, lane = tid & 31, wave = tid >> 5;
  const int row0 = blockIdx.y * 128, col0 = blockIdx.x * 64;
  const int m0 = wave * 16, mr = lane & 15;
  f32x8 acc[4];
  for (int nt = 0; nt < 4; ++nt)
    for (int j = 0; j < 8; ++j) acc[nt][j] = 0.f;

  for (int kt = 0; kt < DINNER / 32; ++kt) {
    const int k0 = kt * 32;
    // stage A via async global->LDS b128 copies (tracked by ASYNCcnt)
    for (int i = 0; i < 2; ++i) {
      int idx = tid + i * 256;                 // 512 chunks of 8 bf16
      int r = idx >> 2, c8 = idx & 3;
      const __bf16* gp = &Ag[(size_t)(row0 + r) * DINNER + k0 + c8 * 8];
      unsigned lds_addr = (unsigned)(uintptr_t)(&sA[r][c8 * 8]);
      unsigned long long gaddr = (unsigned long long)(uintptr_t)gp;
      asm volatile("global_load_async_to_lds_b128 %0, %1, off"
                   :: "v"(lds_addr), "v"(gaddr) : "memory");
    }
    // stage B: b128 loads, transposed store [N][K]
    {
      int r = tid >> 3, c8 = tid & 7;
      bf16x8 wv = *(const bf16x8*)&Wb[(size_t)(k0 + r) * DMODEL + col0 + c8 * 8];
      for (int j = 0; j < 8; ++j) sBt[c8 * 8 + j][r] = wv[j];
    }
    if (kt + 1 < DINNER / 32)
      __builtin_prefetch(&Ag[(size_t)(row0 + (tid >> 1)) * DINNER + k0 + 32], 0, 1);
#if defined(__has_builtin) && __has_builtin(__builtin_amdgcn_s_wait_asynccnt)
    __builtin_amdgcn_s_wait_asynccnt(0);
#else
    asm volatile("s_wait_asynccnt 0x0" ::: "memory");
#endif
    __syncthreads();

    bf16x16 af = ldsFragA(&sA[m0 + mr][0], lane);
    for (int nt = 0; nt < 4; ++nt) {
      bf16x16 bfr = ldsFragB(&sBt[nt * 16 + mr][0], lane);
      acc[nt] = __builtin_amdgcn_wmma_f32_16x16x32_bf16(false, af, false, bfr,
                                                        (short)0, acc[nt], false, false);
    }
    __syncthreads();
  }
  const int mrow = m0 + 8 * (lane >> 4);
  for (int nt = 0; nt < 4; ++nt)
    for (int j = 0; j < 8; ++j)
      Y[(size_t)(row0 + mrow + j) * DMODEL + col0 + nt * 16 + mr] = acc[nt][j];
}

// ---------------- host launcher ----------------
extern "C" void kernel_launch(void* const* d_in, const int* in_sizes, int n_in,
                              void* d_out, int out_size, void* d_ws, size_t ws_size,
                              hipStream_t stream) {
  const float* x       = (const float*)d_in[0];
  const float* W_in    = (const float*)d_in[1];
  const float* W_conv  = (const float*)d_in[2];
  const float* b_conv  = (const float*)d_in[3];
  const float* dt_bias = (const float*)d_in[4];
  const float* A_log   = (const float*)d_in[5];
  const float* D_par   = (const float*)d_in[6];
  const float* gamma   = (const float*)d_in[7];
  const float* beta    = (const float*)d_in[8];
  const float* W_out   = (const float*)d_in[9];
  (void)in_sizes; (void)n_in; (void)out_size; (void)ws_size;

  char* ws = (char*)d_ws;
  // fixed 256B-aligned workspace layout (~496 MB total)
  float*  zx      = (float*) (ws + 0);          // TOK x 1600 fp32      (209,715,200 B)
  __bf16* WinBf   = (__bf16*)(ws + 209715200);  // 384 x 1600 bf16      (  1,228,800 B)
  __bf16* WoutBf  = (__bf16*)(ws + 210944000);  // 768 x 384 bf16       (    589,824 B)
  float*  xconv   = (float*) (ws + 211533824);  // TOK x 800 fp32       (104,857,600 B)
  float*  dtBuf   = (float*) (ws + 316391424);  // TOK x 12 fp32        (  1,572,864 B)
  float*  acumBuf = (float*) (ws + 317964288);  // 6144 x 64 fp32       (  1,572,864 B)
  float*  chunkSt = (float*) (ws + 319537152);  // 6144 x 64 x 16 fp32  ( 25,165,824 B)
  float*  Yssd    = (float*) (ws + 344702976);  // TOK x 768 fp32       (100,663,296 B)
  __bf16* yg      = (__bf16*)(ws + 445366272);  // TOK x 768 bf16       ( 50,331,648 B)
  float*  out     = (float*)d_out;

  k_prep_win <<<(DMODEL * DPROJP + 255) / 256, 256, 0, stream>>>(W_in, WinBf);
  k_prep_wout<<<(DINNER * DMODEL + 255) / 256, 256, 0, stream>>>(W_out, WoutBf);

  dim3 g1(DPROJP / 64, TOK / 128);
  k_gemm_inproj<<<g1, 256, 0, stream>>>(x, WinBf, zx);

  k_dt<<<(TOK * NHEADS + 255) / 256, 256, 0, stream>>>(zx, dt_bias, dtBuf);
  k_conv<<<(BATCH * HW * HW * CONVDIM + 255) / 256, 256, 0, stream>>>(zx, W_conv, b_conv, xconv);

  dim3 g2(NCHUNK, NHEADS, BATCH);
  k_ssd_intra<<<g2, 128, 0, stream>>>(xconv, dtBuf, A_log, D_par, Yssd, chunkSt, acumBuf);

  dim3 g3(NHEADS, BATCH);
  k_ssd_inter<<<g3, 256, 0, stream>>>(xconv, acumBuf, chunkSt, Yssd);

  k_norm_gate<<<TOK, 256, 0, stream>>>(Yssd, zx, gamma, beta, yg);

  dim3 g4(DMODEL / 64, TOK / 128);
  k_gemm_outproj<<<g4, 256, 0, stream>>>(yg, WoutBf, out);
}
